// VectorQuantizer_85401129713684
// MI455X (gfx1250) — matrix operand order repaired and checked by reading, not compiled, not run on hardware
//
#include <hip/hip_runtime.h>

// CDNA5 / gfx1250 VQ-VAE vector quantizer, bf16x3 error-compensated GEMM.
// N=32768 rows (D=64) vs K=1024 codebook entries.
// z.c computed as hi*hi + lo*hi + hi*lo with v_wmma_f32_16x16x32_bf16
// (fp32 accumulate): ~fp32 argmin fidelity at 8x FLOP/instruction vs fp32 WMMA.
// Whole codebook (hi+lo planes = 256KB) staged once per WG into LDS in
// B-fragment order (32B contiguous per lane -> ds_load_b128).
// 512 threads/WG (16 waves); two independent WMMA accumulator chains per
// tile; codebook norms stored transposed so one ds_load_b128 covers 4 tiles.

typedef __attribute__((ext_vector_type(16))) __bf16        v16bf;
typedef __attribute__((ext_vector_type(8)))  float         v8f;
typedef __attribute__((ext_vector_type(8)))  unsigned int  v8u;

#define THREADS_PER_WG  512
#define WAVES_PER_WG    16
#define WG_COUNT        128          // 128 WG * 16 waves * 16 rows = 32768
#define NUMEL           2097152.0f   // 32*64*32*32

// LDS layout (float offsets).
// B region: 64 tiles * 4 blocks (hi/lo x kb0/kb1) * 1024 bytes = 256KB.
//   block(t,p,kbi) = t*4 + p*2 + kbi ; lane slot = 32 bytes at lane*32.
// CBN region: transposed norms cbn[e][t] with padded row stride 68 floats.
#define CBN_OFF         65536        // 16*68 = 1088 floats
#define CBN_STRIDE      68
#define IDX_OFF         66624        // 256 ints: per-wave argmin indices
#define LOSS_OFF        66880        // 16 floats: per-wave loss partials
#define SMEM_FLOATS     66896        // 267584 bytes total (<= 320KB WGP LDS)

__device__ __forceinline__ unsigned bf16_bits(float x) {
    unsigned u = __float_as_uint(x);
    return (u + 0x7FFFu + ((u >> 16) & 1u)) >> 16;   // round-to-nearest-even
}
__device__ __forceinline__ float bf16_val(unsigned b) {
    return __uint_as_float(b << 16);
}

__global__ void __launch_bounds__(THREADS_PER_WG, 1)
vq_main_kernel(const float* __restrict__ z, const float* __restrict__ cb,
               float* __restrict__ out, float* __restrict__ partials)
{
    extern __shared__ float smem[];
    unsigned* smemu = reinterpret_cast<unsigned*>(smem);

    const int tid  = threadIdx.x;
    const int wave = tid >> 5;
    const int lane = tid & 31;
    const int half = lane >> 4;   // 0: lanes 0-15, 1: lanes 16-31
    const int lrow = lane & 15;

    // ---- Prologue 1: codebook -> hi/lo bf16 planes, pre-swizzled into the
    // contiguous-k B order this kernel uses (A matches). B (32x16): lanes 0-15 hold K=0..15 of
    // the block, lanes 16-31 hold K=16..31; each lane's 16 bf16 are the 16
    // CONSECUTIVE dims cb[col][K0..K0+15] -> one 32B contiguous slot per lane.
    const float4* cb4 = reinterpret_cast<const float4*>(cb);
    for (int j = 0; j < 32; ++j) {
        int g = j * THREADS_PER_WG + tid;   // float4 index 0..16383 (coalesced)
        float4 v = cb4[g];
        int entry = g >> 4;             // codebook row
        int t   = entry >> 4;           // tile
        int col = entry & 15;           // column within tile
        int k4  = (g & 15) * 4;         // first dim of this float4
        int kbi = k4 >> 5;              // K-block (0: dims 0-31, 1: dims 32-63)
        int sub = k4 & 31;
        int lane_slot = ((sub >> 4) << 4) + col;   // which lane reads this
        int eoff = sub & 15;                        // element offset in slot
        unsigned h0 = bf16_bits(v.x), h1 = bf16_bits(v.y),
                 h2 = bf16_bits(v.z), h3 = bf16_bits(v.w);
        unsigned l0 = bf16_bits(v.x - bf16_val(h0));
        unsigned l1 = bf16_bits(v.y - bf16_val(h1));
        unsigned l2 = bf16_bits(v.z - bf16_val(h2));
        unsigned l3 = bf16_bits(v.w - bf16_val(h3));
        int hiu = ((t * 4 + kbi)     * 1024 + lane_slot * 32 + eoff * 2) >> 2;
        int lou = ((t * 4 + 2 + kbi) * 1024 + lane_slot * 32 + eoff * 2) >> 2;
        smemu[hiu]     = h0 | (h1 << 16);
        smemu[hiu + 1] = h2 | (h3 << 16);
        smemu[lou]     = l0 | (l1 << 16);
        smemu[lou + 1] = l2 | (l3 << 16);
    }
    // ---- Prologue 2: exact fp32 codebook norms, stored TRANSPOSED:
    // smem[CBN_OFF + e*68 + t] = ||cb[t*16+e]||^2  (padded stride: the 16
    // replicated lanes map to distinct bank groups; 4 tiles per b128 load).
    for (int q = 0; q < 2; ++q) {
        int eg = q * THREADS_PER_WG + tid;
        const float4* row = cb4 + eg * 16;
        float s = 0.f;
        #pragma unroll
        for (int i = 0; i < 16; ++i) {
            float4 v = row[i];
            s += v.x * v.x + v.y * v.y + v.z * v.z + v.w * v.w;
        }
        smem[CBN_OFF + (eg & 15) * CBN_STRIDE + (eg >> 4)] = s;
    }
    __syncthreads();

    // ---- Row block for this wave: 16 consecutive z_flat rows (same image).
    const int n0   = (blockIdx.x * WAVES_PER_WG + wave) * 16;
    const int bimg = n0 >> 10;
    const int np   = n0 & 1023;
    const float* zb = z + (size_t)bimg * 65536 + np;   // + c*1024 + row

    // ---- Build register-resident A fragments (hi/lo, K-blocks 0 and 32).
    // A (16x32 bf16): lane M = lane&15; lanes<16 hold K {0..15},
    // lanes>=16 hold K {16..31}; element e -> K = e + half*16.
    // z is (B,C,H,W): for fixed dim c the 16 rows are contiguous -> coalesced.
    v8u ahu[2], alu[2];
    #pragma unroll
    for (int kbi = 0; kbi < 2; ++kbi) {
        #pragma unroll
        for (int jj = 0; jj < 8; ++jj) {
            int e0 = 2 * jj, e1 = 2 * jj + 1;
            int c0 = kbi * 32 + e0 + half * 16;
            int c1 = kbi * 32 + e1 + half * 16;
            float f0 = zb[(size_t)c0 * 1024 + lrow];
            float f1 = zb[(size_t)c1 * 1024 + lrow];
            unsigned h0 = bf16_bits(f0), h1 = bf16_bits(f1);
            unsigned l0 = bf16_bits(f0 - bf16_val(h0));
            unsigned l1 = bf16_bits(f1 - bf16_val(h1));
            ahu[kbi][jj] = h0 | (h1 << 16);
            alu[kbi][jj] = l0 | (l1 << 16);
        }
    }
    const v16bf ah0 = __builtin_bit_cast(v16bf, ahu[0]);
    const v16bf ah1 = __builtin_bit_cast(v16bf, ahu[1]);
    const v16bf al0 = __builtin_bit_cast(v16bf, alu[0]);
    const v16bf al1 = __builtin_bit_cast(v16bf, alu[1]);

    float minv[8];
    int   mini[8];
    #pragma unroll
    for (int r = 0; r < 8; ++r) { minv[r] = 3.4e38f; mini[r] = 0; }

    // ---- Main loop: 16 groups of 4 tiles. One b128 fetches the 4 norms for
    // the group; per tile: 8x ds_load_b128 + 6 bf16 WMMAs in two independent
    // 3-deep accumulator chains.
    for (int tg = 0; tg < 16; ++tg) {
        float4 cbn4 = *reinterpret_cast<const float4*>(
            &smem[CBN_OFF + lrow * CBN_STRIDE + tg * 4]);
        const float cbnarr[4] = {cbn4.x, cbn4.y, cbn4.z, cbn4.w};
        #pragma unroll
        for (int ti = 0; ti < 4; ++ti) {
            const int t = tg * 4 + ti;
            const v8u* bp = reinterpret_cast<const v8u*>(
                reinterpret_cast<const char*>(smem) + (size_t)t * 4096);
            v8u bh0u = bp[lane];          // hi, K 0-31
            v8u bh1u = bp[32 + lane];     // hi, K 32-63
            v8u bl0u = bp[64 + lane];     // lo, K 0-31
            v8u bl1u = bp[96 + lane];     // lo, K 32-63
            v16bf bh0 = __builtin_bit_cast(v16bf, bh0u);
            v16bf bh1 = __builtin_bit_cast(v16bf, bh1u);
            v16bf bl0 = __builtin_bit_cast(v16bf, bl0u);
            v16bf bl1 = __builtin_bit_cast(v16bf, bl1u);

            v8f acc0 = {0.f, 0.f, 0.f, 0.f, 0.f, 0.f, 0.f, 0.f};
            v8f acc1 = {0.f, 0.f, 0.f, 0.f, 0.f, 0.f, 0.f, 0.f};
            acc0 = __builtin_amdgcn_wmma_f32_16x16x32_bf16(false, ah0, false, bh0, (short)0, acc0, false, false);
            acc1 = __builtin_amdgcn_wmma_f32_16x16x32_bf16(false, ah1, false, bh1, (short)0, acc1, false, false);
            acc0 = __builtin_amdgcn_wmma_f32_16x16x32_bf16(false, al0, false, bh0, (short)0, acc0, false, false);
            acc1 = __builtin_amdgcn_wmma_f32_16x16x32_bf16(false, al1, false, bh1, (short)0, acc1, false, false);
            acc0 = __builtin_amdgcn_wmma_f32_16x16x32_bf16(false, ah0, false, bl0, (short)0, acc0, false, false);
            acc1 = __builtin_amdgcn_wmma_f32_16x16x32_bf16(false, ah1, false, bl1, (short)0, acc1, false, false);

            // score = ||c||^2 - 2*z.c  (||z||^2 is argmin-invariant)
            const int col = t * 16 + lrow;
            const float cbn = cbnarr[ti];
            #pragma unroll
            for (int r = 0; r < 8; ++r) {
                float v = cbn - 2.0f * (acc0[r] + acc1[r]);
                if (v < minv[r]) { minv[r] = v; mini[r] = col; }
            }
        }
    }

    // ---- Argmin across the 16 lanes of each half (C/D layout: VGPR r holds
    // row r for lanes 0-15 and row r+8 for lanes 16-31).
    int* idxs = reinterpret_cast<int*>(&smem[IDX_OFF]);
    #pragma unroll
    for (int r = 0; r < 8; ++r) {
        float v = minv[r];
        int   i = mini[r];
        #pragma unroll
        for (int off = 8; off >= 1; off >>= 1) {
            float vo = __shfl_xor(v, off, 32);
            int   io = __shfl_xor(i, off, 32);
            if (vo < v || (vo == v && io < i)) { v = vo; i = io; }
        }
        if (lrow == 0) idxs[wave * 16 + half * 8 + r] = i;
    }
    __syncthreads();

    // ---- Gather codebook rows, write zq (coalesced), accumulate loss.
    const int myk = idxs[wave * 16 + lrow];
    float* outb = out + (size_t)bimg * 65536 + np;
    float lsum = 0.f;
    for (int cc = 0; cc < 32; ++cc) {
        int c = cc * 2 + half;
        float q  = cb[(size_t)myk * 64 + c];
        float zv = zb[(size_t)c * 1024 + lrow];
        outb[(size_t)c * 1024 + lrow] = q;
        float d = q - zv;
        lsum += d * d;
    }
    #pragma unroll
    for (int off = 16; off >= 1; off >>= 1)
        lsum += __shfl_xor(lsum, off, 32);
    if (lane == 0) smem[LOSS_OFF + wave] = lsum;
    __syncthreads();
    if (tid == 0) {
        float s = 0.f;
        #pragma unroll
        for (int w = 0; w < WAVES_PER_WG; ++w) s += smem[LOSS_OFF + w];
        partials[blockIdx.x] = s;
    }
}

// Deterministic final reduction of 128 per-workgroup partials into the loss.
__global__ void __launch_bounds__(128)
vq_loss_kernel(const float* __restrict__ partials, float* __restrict__ out)
{
    __shared__ float red[128];
    int tid = threadIdx.x;
    red[tid] = partials[tid];
    __syncthreads();
    for (int s = 64; s > 0; s >>= 1) {
        if (tid < s) red[tid] += red[tid + s];
        __syncthreads();
    }
    // loss = (1 + 0.25) * mean((zq - z)^2)
    if (tid == 0) out[2097152] = red[0] * (1.25f / NUMEL);
}

extern "C" void kernel_launch(void* const* d_in, const int* in_sizes, int n_in,
                              void* d_out, int out_size, void* d_ws, size_t ws_size,
                              hipStream_t stream)
{
    (void)in_sizes; (void)n_in; (void)out_size; (void)ws_size;
    const float* z  = (const float*)d_in[0];
    const float* cb = (const float*)d_in[1];
    float* out      = (float*)d_out;
    float* partials = (float*)d_ws;   // 128 floats of scratch

    vq_main_kernel<<<WG_COUNT, THREADS_PER_WG, SMEM_FLOATS * sizeof(float), stream>>>(
        z, cb, out, partials);
    vq_loss_kernel<<<1, 128, 0, stream>>>(partials, out);
}
